// MultiHeadAttention_3427383902664
// MI455X (gfx1250) — matrix operand order repaired and checked by reading, not compiled, not run on hardware
//
#include <hip/hip_runtime.h>
#include <hip/hip_bf16.h>

// ---------------------------------------------------------------------------
// MHA on gfx1250 via v_wmma_f32_16x16x32_bf16.
// B=4, S=2048, D=1024, H=16, Hd=64.  Compute-bound (137 GFLOP vs ~150MB HBM):
// all matmuls run through WMMA (bf16 operands, f32 accum), flash-attention
// online softmax (no S x S materialization).  Waves carry 32x64 register
// tiles so each B fragment feeds two WMMAs (8 WMMA per 12 b128 loads).
// ---------------------------------------------------------------------------

typedef __bf16 bf16;
typedef __attribute__((ext_vector_type(16))) __bf16 v16bf;
typedef __attribute__((ext_vector_type(8)))  __bf16 v8bf;
typedef __attribute__((ext_vector_type(4)))  __bf16 v4bf;
typedef __attribute__((ext_vector_type(8)))  float  v8f;

#define D_MODEL 1024
#define SEQ     2048
#define BATCH   4
#define HEADS   16
#define HDIM    64
#define MROWS   (BATCH * SEQ)       // 8192

static __device__ inline v16bf pack16(v8bf lo, v8bf hi) {
  return __builtin_shufflevector(lo, hi, 0,1,2,3,4,5,6,7,8,9,10,11,12,13,14,15);
}

static __device__ inline v8f wmma_bf16(v16bf a, v16bf b, v8f c) {
  // D = A(16x32) * B(32x16) + C,  f32 accumulate
  return __builtin_amdgcn_wmma_f32_16x16x32_bf16(false, a, false, b,
                                                 (short)0, c, false, false);
}

// ---------------------------------------------------------------------------
// fp32 -> bf16 conversion (x4 vectorized; all sizes divisible by 4)
// ---------------------------------------------------------------------------
__global__ void cvt_f32_bf16(const float* __restrict__ in,
                             bf16* __restrict__ out, int n) {
  int i = (blockIdx.x * blockDim.x + threadIdx.x) * 4;
  if (i < n) {
    float4 f = *(const float4*)(in + i);
    v4bf o;
    o[0] = (bf16)f.x; o[1] = (bf16)f.y; o[2] = (bf16)f.z; o[3] = (bf16)f.w;
    *(v4bf*)(out + i) = o;
  }
}

// ---------------------------------------------------------------------------
// GEMM:  out[m,n] = sum_k A[m,k] * W[n,k] + bias[n]
//   A: [M,K] bf16 row-major, W: [N,K] bf16 row-major (torch Linear layout)
// One wave computes a 32(M) x 64(N) tile: 2 A-fragments x 4 B-fragments,
// 8 accumulators, 8 WMMA per 32-deep K step.  Block = 8 waves arranged
// 2(M) x 4(N) -> block tile 64 x 256.
// MODE 0: store bf16 as [bh][s][hd]      (Q / K layout)
// MODE 1: store bf16 as [bh][hd][s]      (V transposed for PV B-fragments)
// MODE 2: store fp32 as [m][n]           (final output)
// ---------------------------------------------------------------------------
template <int MODE>
__global__ __launch_bounds__(256)
void gemm_bf16_wmma(const bf16* __restrict__ A, const bf16* __restrict__ W,
                    const float* __restrict__ bias, void* __restrict__ outp,
                    int M, int N, int K) {
  const int lane = threadIdx.x & 31;
  const int wave = threadIdx.x >> 5;
  const int lc   = lane & 15;      // column / row-in-group selector
  const int hi   = lane >> 4;      // lane half
  const int m0   = blockIdx.y * 64  + (wave & 1) * 32;
  const int n0   = blockIdx.x * 256 + (wave >> 1) * 64;

  v8f acc[2][4];
#pragma unroll
  for (int rb = 0; rb < 2; ++rb)
#pragma unroll
    for (int t = 0; t < 4; ++t) acc[rb][t] = {};

  const bf16* arow0 = A + (size_t)(m0 + lc) * K;        // rows m0..m0+15
  const bf16* arow1 = A + (size_t)(m0 + 16 + lc) * K;   // rows m0+16..m0+31
  const bf16* brow  = W + (size_t)(n0 + lc) * K;        // cols n0+t*16+lc

  for (int k = 0; k < K; k += 32) {
    // prefetch next tiles into cache while WMMAs run (global_prefetch_b8)
    __builtin_prefetch(arow0 + k + 256, 0, 0);
    __builtin_prefetch(brow  + k + 256, 0, 0);

    // A fragments (16x32): two 16B chunks each, K offsets {hi*8, 16+hi*8}
    v16bf a0 = pack16(*(const v8bf*)(arow0 + k + hi * 8),
                      *(const v8bf*)(arow0 + k + 16 + hi * 8));
    v16bf a1 = pack16(*(const v8bf*)(arow1 + k + hi * 8),
                      *(const v8bf*)(arow1 + k + 16 + hi * 8));
#pragma unroll
    for (int t = 0; t < 4; ++t) {
      // B fragment (32x16): two 16B chunks, K offsets {hi*8, 16+hi*8}
      v16bf b = pack16(*(const v8bf*)(brow + (size_t)t * 16 * K + k + hi * 8),
                       *(const v8bf*)(brow + (size_t)t * 16 * K + k + 16 + hi * 8));
      acc[0][t] = wmma_bf16(a0, b, acc[0][t]);
      acc[1][t] = wmma_bf16(a1, b, acc[1][t]);
    }
  }

#pragma unroll
  for (int rb = 0; rb < 2; ++rb) {
#pragma unroll
    for (int t = 0; t < 4; ++t) {
      const int n = n0 + t * 16 + lc;
      const float bv = bias[n];
#pragma unroll
      for (int r = 0; r < 8; ++r) {
        const int m = m0 + rb * 16 + r + hi * 8;     // C/D layout row
        const float v = acc[rb][t][r] + bv;
        if (MODE == 2) {
          ((float*)outp)[(size_t)m * N + n] = v;
        } else {
          const int bb = m >> 11, s = m & 2047;      // m = b*2048 + s
          const int h  = n >> 6,  hd = n & 63;       // n = h*64 + hd
          size_t idx;
          if (MODE == 0)
            idx = ((size_t)(bb * HEADS + h) * SEQ + s) * HDIM + hd;
          else
            idx = ((size_t)(bb * HEADS + h) * HDIM + hd) * SEQ + s;
          ((bf16*)outp)[idx] = (bf16)v;
        }
      }
    }
  }
}

// ---------------------------------------------------------------------------
// Flash attention: one wave handles a 32-row query tile of one (b,h);
// K / V fragments are reused across both 16-row blocks.
// Per 32-column key chunk:
//   scores (32x32)  : 8 WMMA   (Q 32x64  x  K^T 64x32)
//   online softmax  : shfl_xor reductions inside 16-lane C-fragment groups
//   P staged to LDS : C-fragment -> A-fragment relayout (2KB/wave)
//   context += P*V  : 8 WMMA   (P 32x32  x  V 32x64, V pre-transposed)
// ---------------------------------------------------------------------------
__global__ __launch_bounds__(256)
void flash_attn_wmma(const bf16* __restrict__ Q,   // [bh][s][hd]
                     const bf16* __restrict__ Km,  // [bh][s][hd]
                     const bf16* __restrict__ Vt,  // [bh][hd][s]
                     bf16* __restrict__ Ctx) {     // [b*S][D]
  __shared__ __align__(32) bf16 lds[8 * 2 * 16 * 32];   // 2KB per wave

  const int lane  = threadIdx.x & 31;
  const int wave  = threadIdx.x >> 5;
  const int lc    = lane & 15;
  const int hi    = lane >> 4;
  const int bh    = blockIdx.y;
  const int qtile = blockIdx.x * 8 + wave;              // 32-row tile index
  const int bq    = bh >> 4, hq = bh & 15;

  bf16* pl = lds + wave * 1024;                         // two 16x32 P tiles

  // Q A-fragments: 2 row-blocks x (hd 0..31, hd 32..63)
  v16bf qa[2][2];
#pragma unroll
  for (int rb = 0; rb < 2; ++rb) {
    const bf16* qrow = Q + ((size_t)bh * SEQ + qtile * 32 + rb * 16 + lc) * HDIM;
    qa[rb][0] = pack16(*(const v8bf*)(qrow + hi * 8),
                       *(const v8bf*)(qrow + 16 + hi * 8));
    qa[rb][1] = pack16(*(const v8bf*)(qrow + 32 + hi * 8),
                       *(const v8bf*)(qrow + 48 + hi * 8));
  }

  float mrow[2][8], lrow[2][8];
  v8f acc[2][4];
#pragma unroll
  for (int rb = 0; rb < 2; ++rb) {
#pragma unroll
    for (int r = 0; r < 8; ++r) { mrow[rb][r] = -1e30f; lrow[rb][r] = 0.f; }
#pragma unroll
    for (int t = 0; t < 4; ++t) acc[rb][t] = {};
  }

  const float scale = 0.125f;  // 1/sqrt(64)
  const bf16* kbase = Km + (size_t)bh * SEQ * HDIM;
  const bf16* vbase = Vt + (size_t)bh * HDIM * SEQ;

  for (int c = 0; c < SEQ; c += 32) {
    // ---- K fragments for key columns [c, c+32), shared by both row-blocks --
    const bf16* kr0 = kbase + (size_t)(c + lc) * HDIM;
    const bf16* kr1 = kbase + (size_t)(c + 16 + lc) * HDIM;
    v16bf b00 = pack16(*(const v8bf*)(kr0 + hi * 8), *(const v8bf*)(kr0 + 16 + hi * 8));
    v16bf b01 = pack16(*(const v8bf*)(kr0 + 32 + hi * 8), *(const v8bf*)(kr0 + 48 + hi * 8));
    v16bf b10 = pack16(*(const v8bf*)(kr1 + hi * 8), *(const v8bf*)(kr1 + 16 + hi * 8));
    v16bf b11 = pack16(*(const v8bf*)(kr1 + 32 + hi * 8), *(const v8bf*)(kr1 + 48 + hi * 8));

#pragma unroll
    for (int rb = 0; rb < 2; ++rb) {
      v8f s0 = {}, s1 = {};
      s0 = wmma_bf16(qa[rb][0], b00, s0);
      s0 = wmma_bf16(qa[rb][1], b01, s0);
      s1 = wmma_bf16(qa[rb][0], b10, s1);
      s1 = wmma_bf16(qa[rb][1], b11, s1);

      // ---- online softmax across the 32 columns ----
#pragma unroll
      for (int r = 0; r < 8; ++r) {
        float a0 = s0[r] * scale, a1 = s1[r] * scale;
        float mx = fmaxf(a0, a1);
#pragma unroll
        for (int off = 8; off > 0; off >>= 1)
          mx = fmaxf(mx, __shfl_xor(mx, off, 16));
        const float mnew = fmaxf(mrow[rb][r], mx);
        const float p0 = __expf(a0 - mnew);
        const float p1 = __expf(a1 - mnew);
        float rs = p0 + p1;
#pragma unroll
        for (int off = 8; off > 0; off >>= 1)
          rs += __shfl_xor(rs, off, 16);
        const float f = __expf(mrow[rb][r] - mnew);
        lrow[rb][r] = lrow[rb][r] * f + rs;
        mrow[rb][r] = mnew;
#pragma unroll
        for (int t = 0; t < 4; ++t) acc[rb][t][r] *= f;
        const int prow = r + hi * 8;                  // C layout -> tile row
        pl[rb * 512 + prow * 32 + lc]      = (bf16)p0;
        pl[rb * 512 + prow * 32 + 16 + lc] = (bf16)p1;
      }
    }
    __syncthreads();

    // ---- P (A-fragments) from LDS, context += P * V ----
    v16bf pa0 = pack16(*(const v8bf*)(pl + lc * 32 + hi * 8),
                       *(const v8bf*)(pl + lc * 32 + 16 + hi * 8));
    v16bf pa1 = pack16(*(const v8bf*)(pl + 512 + lc * 32 + hi * 8),
                       *(const v8bf*)(pl + 512 + lc * 32 + 16 + hi * 8));
#pragma unroll
    for (int t = 0; t < 4; ++t) {
      v16bf vb = pack16(*(const v8bf*)(vbase + (size_t)(t * 16 + lc) * SEQ + c + hi * 8),
                        *(const v8bf*)(vbase + (size_t)(t * 16 + lc) * SEQ + c + 16 + hi * 8));
      acc[0][t] = wmma_bf16(pa0, vb, acc[0][t]);
      acc[1][t] = wmma_bf16(pa1, vb, acc[1][t]);
    }
    __syncthreads();
  }

  // ---- normalize and store context as [b*S][D] bf16 ----
#pragma unroll
  for (int rb = 0; rb < 2; ++rb) {
#pragma unroll
    for (int r = 0; r < 8; ++r) {
      const float inv = lrow[rb][r] > 0.f ? 1.f / lrow[rb][r] : 0.f;
      const int s = qtile * 32 + rb * 16 + r + hi * 8;
      const size_t rowbase = ((size_t)bq * SEQ + s) * D_MODEL + hq * HDIM;
#pragma unroll
      for (int t = 0; t < 4; ++t)
        Ctx[rowbase + t * 16 + lc] = (bf16)(acc[rb][t][r] * inv);
    }
  }
}

// ---------------------------------------------------------------------------
extern "C" void kernel_launch(void* const* d_in, const int* in_sizes, int n_in,
                              void* d_out, int out_size, void* d_ws, size_t ws_size,
                              hipStream_t stream) {
  const float* x   = (const float*)d_in[0];
  const float* W_q = (const float*)d_in[1];
  const float* b_q = (const float*)d_in[2];
  const float* W_k = (const float*)d_in[3];
  const float* b_k = (const float*)d_in[4];
  const float* W_v = (const float*)d_in[5];
  const float* b_v = (const float*)d_in[6];
  const float* W_o = (const float*)d_in[7];
  const float* b_o = (const float*)d_in[8];
  float* out = (float*)d_out;

  // Workspace carve-up (bf16 buffers), ~92.3 MB total.
  char* w = (char*)d_ws;
  bf16* xb   = (bf16*)w; w += (size_t)MROWS * D_MODEL * 2;          // 16.8 MB
  bf16* wqb  = (bf16*)w; w += (size_t)D_MODEL * D_MODEL * 2;        //  2 MB
  bf16* wkb  = (bf16*)w; w += (size_t)D_MODEL * D_MODEL * 2;
  bf16* wvb  = (bf16*)w; w += (size_t)D_MODEL * D_MODEL * 2;
  bf16* wob  = (bf16*)w; w += (size_t)D_MODEL * D_MODEL * 2;
  bf16* qb   = (bf16*)w; w += (size_t)BATCH * HEADS * SEQ * HDIM * 2;
  bf16* kb   = (bf16*)w; w += (size_t)BATCH * HEADS * SEQ * HDIM * 2;
  bf16* vt   = (bf16*)w; w += (size_t)BATCH * HEADS * HDIM * SEQ * 2;
  bf16* ctxb = (bf16*)w; w += (size_t)MROWS * D_MODEL * 2;

  // 1) fp32 -> bf16 conversions (x4 vectorized)
  {
    const int nx = MROWS * D_MODEL;
    cvt_f32_bf16<<<(nx / 4 + 255) / 256, 256, 0, stream>>>(x, xb, nx);
    const int nw = D_MODEL * D_MODEL;
    cvt_f32_bf16<<<(nw / 4 + 255) / 256, 256, 0, stream>>>(W_q, wqb, nw);
    cvt_f32_bf16<<<(nw / 4 + 255) / 256, 256, 0, stream>>>(W_k, wkb, nw);
    cvt_f32_bf16<<<(nw / 4 + 255) / 256, 256, 0, stream>>>(W_v, wvb, nw);
    cvt_f32_bf16<<<(nw / 4 + 255) / 256, 256, 0, stream>>>(W_o, wob, nw);
  }

  // 2) Q/K/V projections (M=8192, N=1024, K=1024); block tile 64x256
  dim3 gg(D_MODEL / 256, MROWS / 64);
  gemm_bf16_wmma<0><<<gg, 256, 0, stream>>>(xb, wqb, b_q, (void*)qb,
                                            MROWS, D_MODEL, D_MODEL);
  gemm_bf16_wmma<0><<<gg, 256, 0, stream>>>(xb, wkb, b_k, (void*)kb,
                                            MROWS, D_MODEL, D_MODEL);
  gemm_bf16_wmma<1><<<gg, 256, 0, stream>>>(xb, wvb, b_v, (void*)vt,
                                            MROWS, D_MODEL, D_MODEL);

  // 3) attention: grid (8 qtile-groups of 8 waves x 32 rows, 64 bh)
  dim3 ga(SEQ / (32 * 8), BATCH * HEADS);
  flash_attn_wmma<<<ga, 256, 0, stream>>>(qb, kb, vt, ctxb);

  // 4) output projection -> fp32 d_out
  gemm_bf16_wmma<2><<<gg, 256, 0, stream>>>(ctxb, wob, b_o, (void*)out,
                                            MROWS, D_MODEL, D_MODEL);
}
